// XSAMultiheadAttention_48232482734873
// MI455X (gfx1250) — matrix operand order repaired and checked by reading, not compiled, not run on hardware
//
#include <hip/hip_runtime.h>

#define D_MODEL 1024
#define NHEADS  16
#define DHEAD   64
#define BATCH   2
#define SEQ     2048
#define MTOT    (BATCH*SEQ)   // 4096 rows

typedef __attribute__((ext_vector_type(16))) __bf16          v16bf;
typedef __attribute__((ext_vector_type(8)))  float           v8f;
typedef __attribute__((ext_vector_type(8)))  unsigned short  v8u;
typedef __attribute__((ext_vector_type(16))) unsigned short  v16u;
typedef int v4i __attribute__((vector_size(16)));   // matches async-builtin param

#define GAS __attribute__((address_space(1)))
#define LAS __attribute__((address_space(3)))

// ---- CDNA5 async global->LDS path (guarded; fallback = sync copy) ----
#if __has_builtin(__builtin_amdgcn_global_load_async_to_lds_b128)
#define HAVE_ASYNC 1
#else
#define HAVE_ASYNC 0
#endif

#if HAVE_ASYNC
#if __has_builtin(__builtin_amdgcn_s_wait_asynccnt)
#define WAIT_ASYNC(n) __builtin_amdgcn_s_wait_asynccnt(n)
#else
#define WAIT_ASYNC(n) asm volatile("s_wait_asynccnt %0" :: "i"(n) : "memory")
#endif
#else
#define WAIT_ASYNC(n) ((void)0)
#endif

// ---------- helpers ----------

__device__ __forceinline__ unsigned short f2bf(float f) {
    unsigned int u = __float_as_uint(f);
    u += 0x7FFFu + ((u >> 16) & 1u);          // round-to-nearest-even
    return (unsigned short)(u >> 16);
}

// Build a 16-half WMMA fragment from two 16-byte chunks.
__device__ __forceinline__ v16bf load_frag(const unsigned short* p0,
                                           const unsigned short* p1) {
    v8u lo = *(const v8u*)p0;
    v8u hi = *(const v8u*)p1;
    v16u c = __builtin_shufflevector(lo, hi, 0,1,2,3,4,5,6,7,8,9,10,11,12,13,14,15);
    return __builtin_bit_cast(v16bf, c);
}

// butterfly reductions across the 16-lane half of a wave32
__device__ __forceinline__ float hsum16(float v) {
    v += __shfl_xor(v, 1); v += __shfl_xor(v, 2);
    v += __shfl_xor(v, 4); v += __shfl_xor(v, 8);
    return v;
}
__device__ __forceinline__ float hmax16(float v) {
    v = fmaxf(v, __shfl_xor(v, 1)); v = fmaxf(v, __shfl_xor(v, 2));
    v = fmaxf(v, __shfl_xor(v, 4)); v = fmaxf(v, __shfl_xor(v, 8));
    return v;
}

// ---------- f32 -> bf16 conversion ----------

__global__ void cvt_bf16_kernel(const float* __restrict__ src,
                                unsigned short* __restrict__ dst, int n) {
    int i = blockIdx.x * 256 + threadIdx.x;
    if (i < n) dst[i] = f2bf(src[i]);
}

// ---------- NT GEMM: C(M,N) = A(M,K) * W(N,K)^T, bf16 in / f32 acc ----------
// 8 waves, block tile 128x128; each wave 32(M) x 64(N): every B fragment
// feeds two independent WMMAs (8 wmma per k-step).
// MODE 0: bf16 split-head (B,H,T,dh)            (Q, K projections)
// MODE 1: bf16 transposed (B,H,dh,T) + f32 rows (V projection)
// MODE 2: f32 row-major (B,T,D)                 (output projection)
template<int MODE>
__global__ __launch_bounds__(256)
void gemm_bf16_kernel(const unsigned short* __restrict__ A,
                      const unsigned short* __restrict__ W,
                      unsigned short* __restrict__ out_bf,
                      float* __restrict__ out_f)
{
    const int K = D_MODEL;
    int bx    = blockIdx.x;
    int mbase = (bx >> 3) * 128;
    int nbase = (bx & 7)  * 128;
    int wave  = threadIdx.x >> 5;
    int lane  = threadIdx.x & 31;
    int half  = lane >> 4;
    int l16   = lane & 15;
    int wm    = mbase + (wave & 3) * 32;  // wave's 32 M rows
    int wn    = nbase + (wave >> 2) * 64; // wave's 64 N cols

    const unsigned short* Arow0 = A + (size_t)(wm + l16) * K;
    const unsigned short* Arow1 = A + (size_t)(wm + 16 + l16) * K;
    int a_off = half ? 8 : 0;             // A-layout K offset per lane-half

    v8f vz = {0.f,0.f,0.f,0.f,0.f,0.f,0.f,0.f};
    v8f acc[2][4] = {{vz,vz,vz,vz},{vz,vz,vz,vz}};

    for (int k = 0; k < K; k += 32) {
        v16bf af0 = load_frag(Arow0 + k + a_off, Arow0 + k + a_off + 16);
        v16bf af1 = load_frag(Arow1 + k + a_off, Arow1 + k + a_off + 16);
#pragma unroll
        for (int j = 0; j < 4; ++j) {
            const unsigned short* Wr = W + (size_t)(wn + 16*j + l16) * K + k + half*8;
            v16bf bfr = load_frag(Wr, Wr + 16);
            acc[0][j] = __builtin_amdgcn_wmma_f32_16x16x32_bf16(
                false, af0, false, bfr, (short)0, acc[0][j], false, false);
            acc[1][j] = __builtin_amdgcn_wmma_f32_16x16x32_bf16(
                false, af1, false, bfr, (short)0, acc[1][j], false, false);
        }
    }

#pragma unroll
    for (int mi = 0; mi < 2; ++mi) {
#pragma unroll
        for (int j = 0; j < 4; ++j) {
            int n_g = wn + 16*j + l16;
            int h = n_g >> 6, d = n_g & 63;
#pragma unroll
            for (int r = 0; r < 8; ++r) {
                int m_g = wm + mi*16 + half*8 + r;
                int b = m_g >> 11, t = m_g & (SEQ - 1);
                float v = acc[mi][j][r];
                if (MODE == 0) {
                    out_bf[((size_t)(b*NHEADS + h)*SEQ + t)*DHEAD + d] = f2bf(v);
                } else if (MODE == 1) {
                    out_bf[((size_t)(b*NHEADS + h)*DHEAD + d)*SEQ + t] = f2bf(v);
                    out_f [((size_t)(b*NHEADS + h)*SEQ + t)*DHEAD + d] = v;
                } else {
                    out_f[(size_t)m_g * D_MODEL + n_g] = v;
                }
            }
        }
    }
}

// ---------- flash attention + exclusive output mod ----------
// grid: B*H*(T/64) blocks, 128 threads (4 waves); wave w owns q rows [w*16, w*16+16).
// K/V chunks (8 KB each) staged once per block into LDS via async global->LDS
// copies (double buffered, ASYNCcnt), WMMA B-fragments fed from padded LDS rows.

#define LSTR 72   // LDS row stride in halves (144 B: 16B-aligned, conflict-free)

__device__ __forceinline__ void issue_chunk_copy(
    const unsigned short* __restrict__ Kg,   // K chunk base (contiguous 8 KB)
    const unsigned short* __restrict__ Vg,   // Vt chunk base (64 rows, stride SEQ)
    unsigned short (*Kl)[LSTR], unsigned short (*Vl)[LSTR], int tid)
{
#pragma unroll
    for (int i = 0; i < 4; ++i) {
        int id  = tid * 4 + i;               // 0..511 : (row, 16B-chunk)
        int row = id >> 3, cc = id & 7;
        const unsigned short* gk = Kg + row * DHEAD + cc * 8;
        const unsigned short* gv = Vg + (size_t)row * SEQ + cc * 8;
        unsigned short* lk = &Kl[row][cc * 8];
        unsigned short* lv = &Vl[row][cc * 8];
#if HAVE_ASYNC
        __builtin_amdgcn_global_load_async_to_lds_b128(
            (GAS v4i*)(void*)gk, (LAS v4i*)(void*)lk, 0, 0);
        __builtin_amdgcn_global_load_async_to_lds_b128(
            (GAS v4i*)(void*)gv, (LAS v4i*)(void*)lv, 0, 0);
#else
        *(v8u*)lk = *(const v8u*)gk;
        *(v8u*)lv = *(const v8u*)gv;
#endif
    }
}

__global__ __launch_bounds__(128)
void attn_kernel(const unsigned short* __restrict__ Qb,
                 const unsigned short* __restrict__ Kb,
                 const unsigned short* __restrict__ Vt,
                 const float* __restrict__ Vf,
                 unsigned short* __restrict__ Zb)
{
    __shared__ unsigned short Kbuf[2][64][LSTR];   // 18 KB
    __shared__ unsigned short Vbuf[2][64][LSTR];   // 18 KB
    __shared__ unsigned short Plds[4][16][LSTR];   //  9 KB per-wave P tiles

    int bh = blockIdx.x >> 5;                      // 32 q-tiles per (b,h)
    int qt = blockIdx.x & 31;
    int b  = bh >> 4, h = bh & 15;
    int tid  = threadIdx.x;
    int wave = tid >> 5, lane = tid & 31;
    int half = lane >> 4, l16 = lane & 15;
    int qrow0 = qt * 64 + wave * 16;

    size_t headRM = (size_t)(b*NHEADS + h) * SEQ * DHEAD;  // row-major head base
    size_t headVT = (size_t)(b*NHEADS + h) * DHEAD * SEQ;  // transposed V base
    int a_off = half ? 8 : 0;

    // Q fragments live in registers for the whole kv loop
    const unsigned short* Qrow = Qb + headRM + (size_t)(qrow0 + l16) * DHEAD;
    v16bf qa0 = load_frag(Qrow + a_off,      Qrow + a_off + 16);
    v16bf qa1 = load_frag(Qrow + 32 + a_off, Qrow + 32 + a_off + 16);

    v8f vz = {0.f,0.f,0.f,0.f,0.f,0.f,0.f,0.f};
    v8f yacc[4] = {vz, vz, vz, vz};
    float mrow[8], lrow[8];
#pragma unroll
    for (int r = 0; r < 8; ++r) { mrow[r] = -1e30f; lrow[r] = 0.f; }

    const float sm_scale = 0.125f;                 // 1/sqrt(64)

    // prime double buffer with kv chunk 0
    issue_chunk_copy(Kb + headRM, Vt + headVT, Kbuf[0], Vbuf[0], tid);

    int cur = 0;
    for (int kv = 0; kv < SEQ; kv += 64) {
        int nxt = kv + 64;
        if (nxt < SEQ)
            issue_chunk_copy(Kb + headRM + (size_t)nxt * DHEAD,
                             Vt + headVT + nxt,
                             Kbuf[cur ^ 1], Vbuf[cur ^ 1], tid);
        if (nxt < SEQ) { WAIT_ASYNC(8); } else { WAIT_ASYNC(0); }
        __syncthreads();                           // chunk `kv` visible to all waves

        // S = Q K^T over this 64-col kv chunk (B-fragments from LDS)
        v8f s[4] = {vz, vz, vz, vz};
#pragma unroll
        for (int j = 0; j < 4; ++j) {
            const unsigned short* Kr = &Kbuf[cur][16*j + l16][0];
            v16bf kb0 = load_frag(Kr + half*8,      Kr + half*8 + 16);   // d 0..31
            v16bf kb1 = load_frag(Kr + 32 + half*8, Kr + 48 + half*8);  // d 32..63
            s[j] = __builtin_amdgcn_wmma_f32_16x16x32_bf16(false, qa0, false, kb0, (short)0, s[j], false, false);
            s[j] = __builtin_amdgcn_wmma_f32_16x16x32_bf16(false, qa1, false, kb1, (short)0, s[j], false, false);
        }

        // online softmax (f32, per-row stats replicated across the 16-lane half)
#pragma unroll
        for (int r = 0; r < 8; ++r) {
            float cm = fmaxf(fmaxf(s[0][r], s[1][r]), fmaxf(s[2][r], s[3][r]));
            cm = hmax16(cm) * sm_scale;
            float mnew  = fmaxf(mrow[r], cm);
            float alpha = __expf(mrow[r] - mnew);
            float rs = 0.f;
#pragma unroll
            for (int j = 0; j < 4; ++j) {
                float p = __expf(s[j][r] * sm_scale - mnew);
                s[j][r] = p;
                rs += p;
            }
            rs = hsum16(rs);
            lrow[r] = lrow[r] * alpha + rs;
            mrow[r] = mnew;
#pragma unroll
            for (int j = 0; j < 4; ++j) yacc[j][r] *= alpha;
        }

        // C-layout P -> A-layout via per-wave LDS tile
#pragma unroll
        for (int j = 0; j < 4; ++j) {
            int c = 16*j + l16;
#pragma unroll
            for (int r = 0; r < 8; ++r)
                Plds[wave][half*8 + r][c] = f2bf(s[j][r]);
        }
        const unsigned short* Pr = &Plds[wave][l16][0];
        v16bf pa0 = load_frag(Pr + a_off,      Pr + a_off + 16);      // kv 0..31
        v16bf pa1 = load_frag(Pr + 32 + a_off, Pr + 32 + a_off + 16); // kv 32..63

        // Y += P * V (V B-fragments from LDS)
#pragma unroll
        for (int j = 0; j < 4; ++j) {
            const unsigned short* Vr = &Vbuf[cur][16*j + l16][0];
            v16bf vb0 = load_frag(Vr + half*8,      Vr + half*8 + 16);   // kv 0..31
            v16bf vb1 = load_frag(Vr + 32 + half*8, Vr + 48 + half*8);  // kv 32..63
            yacc[j] = __builtin_amdgcn_wmma_f32_16x16x32_bf16(false, pa0, false, vb0, (short)0, yacc[j], false, false);
            yacc[j] = __builtin_amdgcn_wmma_f32_16x16x32_bf16(false, pa1, false, vb1, (short)0, yacc[j], false, false);
        }

        __syncthreads();                // all waves done reading buf `cur`
        cur ^= 1;
    }

    // epilogue: 1/l, exclusive output mod, merged-head bf16 store
    const float tinyf = 1.17549435e-38f;
    const float epsf  = 1.1920929e-7f;

    float vvals[4][8];
#pragma unroll
    for (int j = 0; j < 4; ++j) {
        int d = 16*j + l16;
#pragma unroll
        for (int r = 0; r < 8; ++r) {
            int t = qrow0 + half*8 + r;
            vvals[j][r] = Vf[headRM + (size_t)t * DHEAD + d];
        }
    }

#pragma unroll
    for (int r = 0; r < 8; ++r) {
        float inv_l = 1.0f / lrow[r];
        float yv[4];
        float ydv = 0.f, vsq = 0.f, ysq = 0.f;
#pragma unroll
        for (int j = 0; j < 4; ++j) {
            float y = yacc[j][r] * inv_l;
            yv[j] = y;
            ydv += y * vvals[j][r];
            vsq += vvals[j][r] * vvals[j][r];
            ysq += y * y;
        }
        ydv = hsum16(ydv); vsq = hsum16(vsq); ysq = hsum16(ysq);

        float scale = (vsq > 0.f) ? (ydv / fmaxf(vsq, tinyf)) : 0.f;
        float z[4]; float zsq = 0.f;
#pragma unroll
        for (int j = 0; j < 4; ++j) { z[j] = yv[j] - scale * vvals[j][r]; zsq += z[j]*z[j]; }
        zsq = hsum16(zsq);

        float refn = fmaxf(sqrtf(ysq), sqrtf(vsq));
        bool kill = (vsq > 0.f) && (sqrtf(zsq) <= epsf * (float)DHEAD * refn);

        int t = qrow0 + half*8 + r;
#pragma unroll
        for (int j = 0; j < 4; ++j) {
            int d = 16*j + l16;
            float zo = kill ? 0.f : z[j];
            Zb[((size_t)(b*SEQ + t))*D_MODEL + h*DHEAD + d] = f2bf(zo);
        }
    }
}

// ---------- launcher ----------

extern "C" void kernel_launch(void* const* d_in, const int* in_sizes, int n_in,
                              void* d_out, int out_size, void* d_ws, size_t ws_size,
                              hipStream_t stream)
{
    const float* x  = (const float*)d_in[0];
    const float* Wq = (const float*)d_in[1];
    const float* Wk = (const float*)d_in[2];
    const float* Wv = (const float*)d_in[3];
    const float* Wo = (const float*)d_in[4];

    char* ws = (char*)d_ws;
    const size_t MB = 1024 * 1024;
    unsigned short* xb  = (unsigned short*)(ws +  0*MB);  //  8 MB  x bf16
    unsigned short* Wqb = (unsigned short*)(ws +  8*MB);  //  2 MB
    unsigned short* Wkb = (unsigned short*)(ws + 10*MB);  //  2 MB
    unsigned short* Wvb = (unsigned short*)(ws + 12*MB);  //  2 MB
    unsigned short* Wob = (unsigned short*)(ws + 14*MB);  //  2 MB
    unsigned short* Qb  = (unsigned short*)(ws + 16*MB);  //  8 MB  (B,H,T,dh) bf16
    unsigned short* Kb  = (unsigned short*)(ws + 24*MB);  //  8 MB  (B,H,T,dh) bf16
    unsigned short* Vt  = (unsigned short*)(ws + 32*MB);  //  8 MB  (B,H,dh,T) bf16
    float*          Vf  = (float*)        (ws + 40*MB);   // 16 MB  (B,H,T,dh) f32
    unsigned short* Zb  = (unsigned short*)(ws + 56*MB);  //  8 MB  (B,T,D) bf16

    const int NX = MTOT * D_MODEL;       // 4194304
    const int NW = D_MODEL * D_MODEL;    // 1048576
    cvt_bf16_kernel<<<(NX + 255)/256, 256, 0, stream>>>(x,  xb,  NX);
    cvt_bf16_kernel<<<(NW + 255)/256, 256, 0, stream>>>(Wq, Wqb, NW);
    cvt_bf16_kernel<<<(NW + 255)/256, 256, 0, stream>>>(Wk, Wkb, NW);
    cvt_bf16_kernel<<<(NW + 255)/256, 256, 0, stream>>>(Wv, Wvb, NW);
    cvt_bf16_kernel<<<(NW + 255)/256, 256, 0, stream>>>(Wo, Wob, NW);

    dim3 gg(256), gb(256);               // (4096/128) x (1024/128) tiles
    gemm_bf16_kernel<0><<<gg, gb, 0, stream>>>(xb, Wqb, Qb, nullptr);
    gemm_bf16_kernel<0><<<gg, gb, 0, stream>>>(xb, Wkb, Kb, nullptr);
    gemm_bf16_kernel<1><<<gg, gb, 0, stream>>>(xb, Wvb, Vt, Vf);

    attn_kernel<<<BATCH*NHEADS*(SEQ/64), 128, 0, stream>>>(Qb, Kb, Vt, Vf, Zb);

    gemm_bf16_kernel<2><<<gg, gb, 0, stream>>>(Zb, Wob, nullptr, (float*)d_out);
}
